// MambaBlock_39943195852866
// MI455X (gfx1250) — matrix operand order repaired and checked by reading, not compiled, not run on hardware
//
#include <hip/hip_runtime.h>
#include <hip/hip_bf16.h>

// ---------------------------------------------------------------------------
// Mamba block forward for MI455X (gfx1250, wave32, WMMA).
// GEMMs: bf16 operands, f32 accumulate via v_wmma_f32_16x16x32_bf16.
// ---------------------------------------------------------------------------

typedef __attribute__((ext_vector_type(16))) __bf16 v16bf;
typedef __attribute__((ext_vector_type(8)))  float  v8f;
typedef __attribute__((ext_vector_type(4)))  __bf16 bf16x4;

#define DM    1024
#define DI    2048
#define NST   16
#define DTR   128
#define MROWS 2048          // B * L
#define LDT   36            // LDS tile row stride (elements) for 32-wide K tiles

// ------------------------------ RMSNorm + bf16 cast ------------------------
__global__ __launch_bounds__(256)
void rmsnorm_cast_kernel(const float* __restrict__ x, const float* __restrict__ w,
                         __bf16* __restrict__ h) {
    __shared__ float red[256];
    const int row = blockIdx.x;
    const float* xr = x + (size_t)row * DM;
    float s = 0.f;
    for (int i = threadIdx.x; i < DM; i += 256) { float v = xr[i]; s += v * v; }
    red[threadIdx.x] = s;
    __syncthreads();
    for (int off = 128; off > 0; off >>= 1) {
        if (threadIdx.x < off) red[threadIdx.x] += red[threadIdx.x + off];
        __syncthreads();
    }
    const float scale = rsqrtf(red[0] * (1.0f / DM) + 1e-6f);
    for (int i = threadIdx.x; i < DM; i += 256)
        h[(size_t)row * DM + i] = (__bf16)(xr[i] * scale * w[i]);
}

// ------------------------------ generic f32 -> bf16 cast -------------------
__global__ __launch_bounds__(256)
void cast_bf16_kernel(const float* __restrict__ s, __bf16* __restrict__ d, int n) {
    int i = blockIdx.x * 256 + threadIdx.x;
    if (i < n) d[i] = (__bf16)s[i];
}

// ------------------------------ bf16 WMMA GEMM -----------------------------
// C[M,N] = A[M,K] * W[N,K]^T (+ residual). A,W bf16 row-major (K contiguous).
// Block: 256 threads (8 waves) -> 128x128 C tile. Wave w: rows [16w,16w+16),
// eight 16x16 N-subtiles, K consumed 32 at a time through LDS.
__global__ __launch_bounds__(256)
void gemm_bf16_wmma_kernel(const __bf16* __restrict__ A, const __bf16* __restrict__ W,
                           float* __restrict__ C, const float* __restrict__ residual,
                           int M, int N, int K) {
    __shared__ __bf16 sA[128 * LDT];
    __shared__ __bf16 sB[128 * LDT];

    const int tid    = threadIdx.x;
    const int wave   = tid >> 5;
    const int lane   = tid & 31;
    const int laneLo = lane & 15;
    const int laneHi = lane >> 4;
    const int mBase  = blockIdx.y * 128;
    const int nBase  = blockIdx.x * 128;

    v8f acc[8];
#pragma unroll
    for (int t = 0; t < 8; ++t) { v8f z = {}; acc[t] = z; }

    // tile-load role: 2 threads per row, each covers 16 of the 32 K columns
    const int rowL  = tid & 127;
    const int kHalf = (tid >> 7) * 16;
    int nRow = nBase + rowL; if (nRow >= N) nRow = N - 1;   // safe clamp for tails

    for (int kb = 0; kb < K; kb += 32) {
        // ---- stage A and W tiles into LDS (8B vector moves) ----
        {
            const bf16x4* gA = (const bf16x4*)(A + (size_t)(mBase + rowL) * K + kb + kHalf);
            bf16x4* dA = (bf16x4*)&sA[rowL * LDT + kHalf];
            dA[0] = gA[0]; dA[1] = gA[1]; dA[2] = gA[2]; dA[3] = gA[3];
            const bf16x4* gB = (const bf16x4*)(W + (size_t)nRow * K + kb + kHalf);
            bf16x4* dB = (bf16x4*)&sB[rowL * LDT + kHalf];
            dB[0] = gB[0]; dB[1] = gB[1]; dB[2] = gB[2]; dB[3] = gB[3];
            if (kb + 32 < K) {  // hint next K tile toward the caches
                __builtin_prefetch(A + (size_t)(mBase + rowL) * K + kb + 32 + kHalf, 0, 0);
                __builtin_prefetch(W + (size_t)nRow * K + kb + 32 + kHalf, 0, 0);
            }
        }
        __syncthreads();

        // ---- gather A fragment (ISA 16-bit A 16x32 layout) ----
        union { v16bf v; unsigned int u[8]; } afrag;
        const int ar = wave * 16 + laneLo;
#pragma unroll
        for (int v = 0; v < 8; ++v) {
            const int k0 = (v < 4) ? (2 * v + 8 * laneHi) : (16 + 2 * (v - 4) + 8 * laneHi);
            afrag.u[v] = *(const unsigned int*)&sA[ar * LDT + k0];
        }

        // ---- 8 N-subtiles: gather B fragment, issue WMMA ----
#pragma unroll
        for (int t = 0; t < 8; ++t) {
            union { v16bf v; unsigned int u[8]; } bfrag;
            const int bn = t * 16 + laneLo;
#pragma unroll
            for (int v = 0; v < 8; ++v) {
                const int k0 = (v < 4) ? (2 * v + 8 * laneHi) : (16 + 2 * (v - 4) + 8 * laneHi);
                bfrag.u[v] = *(const unsigned int*)&sB[bn * LDT + k0];
            }
            acc[t] = __builtin_amdgcn_wmma_f32_16x16x32_bf16(
                false, afrag.v, false, bfrag.v, (short)0, acc[t], false, false);
        }
        __syncthreads();
    }

    // ---- epilogue: C/D layout VGPR r -> M = r + 8*laneHi, N = laneLo ----
#pragma unroll
    for (int t = 0; t < 8; ++t) {
        const int n = nBase + t * 16 + laneLo;
        if (n < N) {
#pragma unroll
            for (int r = 0; r < 8; ++r) {
                const int m = mBase + wave * 16 + r + 8 * laneHi;
                const size_t idx = (size_t)m * N + n;
                float val = acc[t][r];
                if (residual) val += residual[idx];
                C[idx] = val;
            }
        }
    }
}

// --------------- causal depthwise conv (k=4) + bias + SiLU -----------------
// xz row-major [MROWS, 2*DI]; xb is cols [0,DI). Writes f32 and bf16 copies.
__global__ __launch_bounds__(256)
void conv_silu_kernel(const float* __restrict__ xz, const float* __restrict__ cw,
                      const float* __restrict__ cb, float* __restrict__ xc_f,
                      __bf16* __restrict__ xc_bf) {
    const int idx = blockIdx.x * 256 + threadIdx.x;
    if (idx >= MROWS * DI) return;
    const int d = idx & (DI - 1);
    const int m = idx >> 11;
    const int l = m & 1023;
    const int b = m >> 10;
    const float* wr = cw + d * 4;
    float acc = cb[d];
#pragma unroll
    for (int j = 0; j < 4; ++j) {
        const int ls = l - 3 + j;
        if (ls >= 0) acc += wr[j] * xz[((size_t)(b * 1024 + ls)) * (2 * DI) + d];
    }
    const float v = acc / (1.0f + __expf(-acc));
    xc_f[idx]  = v;
    xc_bf[idx] = (__bf16)v;
}

// --------------- slice dt-rank columns of dbl, cast to bf16 ----------------
__global__ __launch_bounds__(256)
void dtin_cast_kernel(const float* __restrict__ dbl, __bf16* __restrict__ dtin) {
    const int idx = blockIdx.x * 256 + threadIdx.x;
    if (idx >= MROWS * DTR) return;
    const int j = idx & (DTR - 1);
    const int m = idx >> 7;
    dtin[idx] = (__bf16)dbl[(size_t)m * (DTR + 2 * NST) + j];
}

// --------------- softplus(dt + b_dt), in place ----------------------------
__global__ __launch_bounds__(256)
void softplus_bias_kernel(float* __restrict__ dt, const float* __restrict__ b_dt) {
    const int idx = blockIdx.x * 256 + threadIdx.x;
    if (idx >= MROWS * DI) return;
    const int d = idx & (DI - 1);
    const float v = dt[idx] + b_dt[d];
    dt[idx] = (v > 20.0f) ? v : log1pf(__expf(v));
}

// --------------- selective scan -------------------------------------------
// One thread per (b,d) channel, 16-state recurrence in registers. B/C rows
// (shared across all d of a batch) staged per 64-step chunk in LDS.
// Writes y = ssm_out + D*x IN PLACE over xc_f (element read before write).
#define SCAN_CHUNK 64
__global__ __launch_bounds__(256)
void scan_kernel(const float* __restrict__ dt, float* __restrict__ xc_y,
                 const float* __restrict__ dbl, const float* __restrict__ A_log,
                 const float* __restrict__ Dvec) {
    __shared__ float lBC[SCAN_CHUNK * 32];   // per row: B[16] then C[16]
    const int tid = threadIdx.x;
    const int b   = blockIdx.x >> 3;                 // DI/256 = 8 chunks per batch
    const int d   = ((blockIdx.x & 7) << 8) + tid;

    float Areg[NST];
#pragma unroll
    for (int n = 0; n < NST; ++n) Areg[n] = -__expf(A_log[(size_t)d * NST + n]);
    const float Dv = Dvec[d];

    float hst[NST];
#pragma unroll
    for (int n = 0; n < NST; ++n) hst[n] = 0.f;

    for (int lc = 0; lc < 1024; lc += SCAN_CHUNK) {
        const int row0 = b * 1024 + lc;
        for (int i = tid; i < SCAN_CHUNK * 32; i += 256) {
            const int r = i >> 5, c = i & 31;
            lBC[i] = dbl[(size_t)(row0 + r) * (DTR + 2 * NST) + DTR + c];
        }
        __syncthreads();
        for (int r = 0; r < SCAN_CHUNK; ++r) {
            const size_t off = (size_t)(row0 + r) * DI + d;
            const float dtv = dt[off];
            const float xv  = xc_y[off];
            const float dbx = dtv * xv;
            const float* Brow = &lBC[r * 32];
            float acc = 0.f;
#pragma unroll
            for (int n = 0; n < NST; ++n) {
                const float dA = __expf(dtv * Areg[n]);
                hst[n] = dA * hst[n] + dbx * Brow[n];
                acc += hst[n] * Brow[16 + n];
            }
            xc_y[off] = acc + Dv * xv;               // y overwrites conv buffer
        }
        __syncthreads();
    }
}

// --------------- gating: yg = y * silu(z), cast to bf16 --------------------
__global__ __launch_bounds__(256)
void gate_cast_kernel(const float* __restrict__ y, const float* __restrict__ xz,
                      __bf16* __restrict__ yg) {
    const int idx = blockIdx.x * 256 + threadIdx.x;
    if (idx >= MROWS * DI) return;
    const int d = idx & (DI - 1);
    const int m = idx >> 11;
    const float z = xz[(size_t)m * (2 * DI) + DI + d];
    const float g = z / (1.0f + __expf(-z));
    yg[idx] = (__bf16)(y[idx] * g);
}

// ---------------------------------------------------------------------------
extern "C" void kernel_launch(void* const* d_in, const int* in_sizes, int n_in,
                              void* d_out, int out_size, void* d_ws, size_t ws_size,
                              hipStream_t stream) {
    const float* x      = (const float*)d_in[0];
    const float* norm_w = (const float*)d_in[1];
    const float* W_in   = (const float*)d_in[2];
    const float* conv_w = (const float*)d_in[3];
    const float* conv_b = (const float*)d_in[4];
    const float* W_xp   = (const float*)d_in[5];
    const float* W_dt   = (const float*)d_in[6];
    const float* b_dt   = (const float*)d_in[7];
    const float* A_log  = (const float*)d_in[8];
    const float* Dvec   = (const float*)d_in[9];
    const float* W_out  = (const float*)d_in[10];
    float* out = (float*)d_out;

    // ---- workspace carve-out (256B aligned) ----
    char* wsb = (char*)d_ws;
    size_t off = 0;
    auto alloc = [&](size_t bytes) -> void* {
        void* p = wsb + off;
        off += (bytes + 255) & ~(size_t)255;
        return p;
    };
    __bf16* h_bf    = (__bf16*)alloc((size_t)MROWS * DM * 2);
    __bf16* Win_bf  = (__bf16*)alloc((size_t)2 * DI * DM * 2);
    __bf16* Wxp_bf  = (__bf16*)alloc((size_t)(DTR + 2 * NST) * DI * 2);
    __bf16* Wdt_bf  = (__bf16*)alloc((size_t)DI * DTR * 2);
    __bf16* Wout_bf = (__bf16*)alloc((size_t)DM * DI * 2);
    float*  xz      = (float*) alloc((size_t)MROWS * 2 * DI * 4);
    float*  xc_f    = (float*) alloc((size_t)MROWS * DI * 4);   // conv out, then y
    __bf16* xc_bf   = (__bf16*)alloc((size_t)MROWS * DI * 2);
    float*  dbl     = (float*) alloc((size_t)MROWS * (DTR + 2 * NST) * 4);
    __bf16* dtin_bf = (__bf16*)alloc((size_t)MROWS * DTR * 2);
    float*  dt_f    = (float*) alloc((size_t)MROWS * DI * 4);
    __bf16* yg_bf   = (__bf16*)alloc((size_t)MROWS * DI * 2);
    (void)ws_size; (void)in_sizes; (void)n_in; (void)out_size;

    // 1) RMSNorm + bf16 cast
    rmsnorm_cast_kernel<<<MROWS, 256, 0, stream>>>(x, norm_w, h_bf);

    // 2) weight casts to bf16
    auto cast = [&](const float* s, __bf16* d, int n) {
        cast_bf16_kernel<<<(n + 255) / 256, 256, 0, stream>>>(s, d, n);
    };
    cast(W_in,  Win_bf,  2 * DI * DM);
    cast(W_xp,  Wxp_bf,  (DTR + 2 * NST) * DI);
    cast(W_dt,  Wdt_bf,  DI * DTR);
    cast(W_out, Wout_bf, DM * DI);

    // 3) in_proj: xz = h @ W_in^T   [2048 x 4096]
    gemm_bf16_wmma_kernel<<<dim3(2 * DI / 128, MROWS / 128), 256, 0, stream>>>(
        h_bf, Win_bf, xz, nullptr, MROWS, 2 * DI, DM);

    // 4) causal depthwise conv + SiLU
    conv_silu_kernel<<<(MROWS * DI) / 256, 256, 0, stream>>>(
        xz, conv_w, conv_b, xc_f, xc_bf);

    // 5) x_proj: dbl = xb @ W_xproj^T   [2048 x 160]
    gemm_bf16_wmma_kernel<<<dim3(2, MROWS / 128), 256, 0, stream>>>(
        xc_bf, Wxp_bf, dbl, nullptr, MROWS, DTR + 2 * NST, DI);

    // 6) slice + cast dt-rank inputs
    dtin_cast_kernel<<<(MROWS * DTR) / 256, 256, 0, stream>>>(dbl, dtin_bf);

    // 7) dt_proj: dt = dtin @ W_dt^T   [2048 x 2048]
    gemm_bf16_wmma_kernel<<<dim3(DI / 128, MROWS / 128), 256, 0, stream>>>(
        dtin_bf, Wdt_bf, dt_f, nullptr, MROWS, DI, DTR);

    // 8) softplus(dt + b_dt)
    softplus_bias_kernel<<<(MROWS * DI) / 256, 256, 0, stream>>>(dt_f, b_dt);

    // 9) selective scan (y written in place over xc_f)
    scan_kernel<<<2 * (DI / 256), 256, 0, stream>>>(dt_f, xc_f, dbl, A_log, Dvec);

    // 10) gating + bf16 cast
    gate_cast_kernel<<<(MROWS * DI) / 256, 256, 0, stream>>>(xc_f, xz, yg_bf);

    // 11) out_proj + residual: out = yg @ W_out^T + x   [2048 x 1024]
    gemm_bf16_wmma_kernel<<<dim3(DM / 128, MROWS / 128), 256, 0, stream>>>(
        yg_bf, Wout_bf, out, x, MROWS, DM, DI);
}
